// S2DTA_12489764897617
// MI455X (gfx1250) — matrix operand order repaired and checked by reading, not compiled, not run on hardware
//
#include <hip/hip_runtime.h>
#include <hip/hip_bf16.h>

typedef __attribute__((ext_vector_type(16))) _Float16 v16h;
typedef __attribute__((ext_vector_type(8)))  float    v8f;

union CFrag { v8f v; float f[8]; };

#define WMMA_F16(a, b, c) \
    __builtin_amdgcn_wmma_f32_16x16x32_f16(false, (a), false, (b), (short)0, (c), false, false)

// ---------------------------------------------------------------------------
// Panel loads: both A and B fragments are pre-packed in f16 so each lane does a
// single aligned 32B vector load (2x global_load_b128), no converts in the loop.
//   A panels: P[((mtile*(K/32) + chunk)*512) + lane*16 + h]
//   B panels: P[((chunk*(N/16) + ntile)*512) + lane*16 + h]
// ---------------------------------------------------------------------------
__device__ __forceinline__ v16h load_ap(const _Float16* __restrict__ P,
                                        int kchunks, int mtile, int chunk) {
    int lane = threadIdx.x & 31;
    return *(const v16h*)(P + (((size_t)mtile * kchunks + chunk) << 9) + (lane << 4));
}
__device__ __forceinline__ v16h load_bp(const _Float16* __restrict__ P,
                                        int ntiles, int chunk, int ntile) {
    int lane = threadIdx.x & 31;
    return *(const v16h*)(P + (((size_t)chunk * ntiles + ntile) << 9) + (lane << 4));
}

// ---------------------------------------------------------------------------
// Pack X [M,K] f32 (row-major, stride K, K%32==0, M%16==0) into A-fragment panels.
// A-layout: lanes 0-15 -> M=lane%16, halves 0..7 = K+0..7, 8..15 = K+16..23;
//           lanes 16-31 -> +8 on the K base.
// ---------------------------------------------------------------------------
__global__ void pack_a_kernel(const float* __restrict__ X, int K, long long total,
                              _Float16* __restrict__ P) {
    long long idx = (long long)blockIdx.x * 256 + threadIdx.x;
    if (idx >= total) return;           // total = M*K
    int h    = (int)(idx & 15);
    int lane = (int)((idx >> 4) & 31);
    long long rest = idx >> 9;
    int kchunks = K >> 5;
    int chunk = (int)(rest % kchunks);
    long long mtile = rest / kchunks;
    long long m = mtile * 16 + (lane & 15);
    int k = chunk * 32 + ((lane >> 4) << 3) + ((h >> 3) << 4) + (h & 7);
    P[idx] = (_Float16)X[m * K + k];
}

// ---------------------------------------------------------------------------
// Pack W [Ksrc,N] f32 into B-fragment panels, K zero-padded to Kpad.
// ---------------------------------------------------------------------------
__global__ void pack_b_kernel(const float* __restrict__ W, int Ksrc, int N, int Kpad,
                              _Float16* __restrict__ P) {
    int idx = blockIdx.x * 256 + threadIdx.x;
    int total = Kpad * N;
    if (idx >= total) return;
    int h    = idx & 15;
    int lane = (idx >> 4) & 31;
    int rest = idx >> 9;
    int ntiles = N >> 4;
    int ntile = rest % ntiles;
    int chunk = rest / ntiles;
    int n = ntile * 16 + (lane & 15);
    int k = chunk * 32 + ((lane >> 4) << 3) + ((h >> 3) << 4) + (h & 7);
    P[idx] = (k < Ksrc) ? (_Float16)W[(size_t)k * N + n] : (_Float16)0.f;
}

// ---------------------------------------------------------------------------
// Zero-padded feature copy: Y[M,Fp] = [ X[M,F] | 0 ]
// ---------------------------------------------------------------------------
__global__ void pad_copy_kernel(const float* __restrict__ X, int F, int Fp, long long total,
                                float* __restrict__ Y) {
    long long idx = (long long)blockIdx.x * blockDim.x + threadIdx.x;
    if (idx >= total) return;
    int f = (int)(idx % Fp);
    long long i = idx / Fp;
    Y[idx] = (f < F) ? X[i * F + f] : 0.f;
}

// ---------------------------------------------------------------------------
// Fused SAGE layer: Y[M,N] = lrelu( Agg@Wl + X@Wr + bias ), all operands packed f16.
// One wave per 16x64 strip: per 32-K chunk -> 2 A loads, 8 B loads, 8 WMMAs.
// ---------------------------------------------------------------------------
__global__ void __launch_bounds__(32)
wmma_sage_kernel(const _Float16* __restrict__ AggP, const _Float16* __restrict__ XP,
                 const _Float16* __restrict__ WlP, const _Float16* __restrict__ WrP,
                 const float* __restrict__ bias, float* __restrict__ Y,
                 int M, int K, int N) {
    int mtile = blockIdx.x;
    int nt0   = blockIdx.y * 4;
    int lane  = threadIdx.x & 31;
    int kchunks = K >> 5;
    int ntiles  = N >> 4;
    v8f acc[4] = {{}, {}, {}, {}};
#pragma unroll 2
    for (int c = 0; c < kchunks; ++c) {
        if (c + 1 < kchunks) {  // gfx1250 global_prefetch_b8: A panels are sequential
            __builtin_prefetch(AggP + (((size_t)mtile * kchunks + c + 1) << 9) + (lane << 4), 0, 1);
            __builtin_prefetch(XP   + (((size_t)mtile * kchunks + c + 1) << 9) + (lane << 4), 0, 1);
        }
        v16h aA = load_ap(AggP, kchunks, mtile, c);
        v16h aX = load_ap(XP,   kchunks, mtile, c);
#pragma unroll
        for (int t = 0; t < 4; ++t) {
            v16h bL = load_bp(WlP, ntiles, c, nt0 + t);
            acc[t] = WMMA_F16(aA, bL, acc[t]);
            v16h bR = load_bp(WrP, ntiles, c, nt0 + t);
            acc[t] = WMMA_F16(aX, bR, acc[t]);
        }
    }
    // C/D layout: VGPR r -> lanes 0-15: M=r, N=lane; lanes 16-31: M=r+8, N=lane-16
    int mb = mtile * 16 + ((lane >> 4) << 3);
#pragma unroll
    for (int t = 0; t < 4; ++t) {
        int n = (nt0 + t) * 16 + (lane & 15);
        float bv = bias[n];
        CFrag c2; c2.v = acc[t];
#pragma unroll
        for (int r = 0; r < 8; ++r) {
            float y = c2.f[r] + bv;
            y = (y > 0.f) ? y : 0.01f * y;
            Y[(size_t)(mb + r) * N + n] = y;
        }
    }
}

// ---------------------------------------------------------------------------
// Linear head: Y[:, ycol:ycol+N] = act( X@W + bias ), packed operands.
// ---------------------------------------------------------------------------
__global__ void __launch_bounds__(32)
wmma_linear_kernel(const _Float16* __restrict__ XP, const _Float16* __restrict__ WP,
                   const float* __restrict__ bias, float* __restrict__ Y,
                   int ldy, int ycol, int M, int K, int N, int do_lrelu) {
    int mtile = blockIdx.x;
    int nt0   = blockIdx.y * 4;
    int lane  = threadIdx.x & 31;
    int kchunks = K >> 5;
    int ntiles  = N >> 4;
    v8f acc[4] = {{}, {}, {}, {}};
#pragma unroll 2
    for (int c = 0; c < kchunks; ++c) {
        v16h a = load_ap(XP, kchunks, mtile, c);
#pragma unroll
        for (int t = 0; t < 4; ++t) {
            v16h b = load_bp(WP, ntiles, c, nt0 + t);
            acc[t] = WMMA_F16(a, b, acc[t]);
        }
    }
    int mb = mtile * 16 + ((lane >> 4) << 3);
#pragma unroll
    for (int t = 0; t < 4; ++t) {
        int n = (nt0 + t) * 16 + (lane & 15);
        float bv = bias[n];
        CFrag c2; c2.v = acc[t];
#pragma unroll
        for (int r = 0; r < 8; ++r) {
            float y = c2.f[r] + bv;
            if (do_lrelu) y = (y > 0.f) ? y : 0.01f * y;
            Y[(size_t)(mb + r) * ldy + ycol + n] = y;
        }
    }
}

// ---------------------------------------------------------------------------
// Conv1d(21->128, k=5, pad=2) fused with ragged gather: one block per packed row.
// ---------------------------------------------------------------------------
__global__ void __launch_bounds__(128)
conv_gather_kernel(const float* __restrict__ seq_x, const float* __restrict__ conv_w,
                   const float* __restrict__ conv_b, const int* __restrict__ prot_gather,
                   float* __restrict__ p0) {
    __shared__ float xs[5][21];
    int i = blockIdx.x;
    int g = prot_gather[i];
    int b = g >> 10;        // LMAX = 1024
    int l = g & 1023;
    int t = threadIdx.x;
    if (t < 105) {
        int tt = t / 21, cc = t % 21;
        int ll = l + tt - 2;
        xs[tt][cc] = (ll >= 0 && ll < 1024)
                   ? seq_x[((size_t)b * 1024 + ll) * 21 + cc] : 0.f;
    }
    __syncthreads();
    int o = threadIdx.x;    // output channel 0..127
    float acc = conv_b[o];
    const float* w = conv_w + (size_t)o * 105;   // [21][5]
#pragma unroll
    for (int c = 0; c < 21; ++c)
#pragma unroll
        for (int tt = 0; tt < 5; ++tt)
            acc += xs[tt][c] * w[c * 5 + tt];
    p0[(size_t)i * 128 + o] = acc;
}

// ---------------------------------------------------------------------------
// Edge scatter-add: Agg[tgt[e]*astride + f] += X[src[e]*F + f]  (float atomics)
// ---------------------------------------------------------------------------
__global__ void scatter_add_kernel(const float* __restrict__ X,
                                   const int* __restrict__ src,
                                   const int* __restrict__ tgt,
                                   long long total, int F, int astride,
                                   float* __restrict__ Agg) {
    long long idx = (long long)blockIdx.x * blockDim.x + threadIdx.x;
    if (idx >= total) return;
    int e = (int)(idx / F);
    int f = (int)(idx % F);
    atomicAdd(&Agg[(size_t)tgt[e] * astride + f], X[(size_t)src[e] * F + f]);
}

// ---------------------------------------------------------------------------
// Segment starts via binary search over a sorted batch vector.
// ---------------------------------------------------------------------------
__global__ void seg_starts_kernel(const int* __restrict__ batch, int n, int nseg,
                                  int* __restrict__ starts) {
    int b = threadIdx.x;
    if (b > nseg) return;
    if (b == nseg) { starts[b] = n; return; }
    int lo = 0, hi = n;
    while (lo < hi) { int mid = (lo + hi) >> 1; if (batch[mid] < b) lo = mid + 1; else hi = mid; }
    starts[b] = lo;
}

// ---------------------------------------------------------------------------
// Segment max pool: Y[b,f] = max over rows [starts[b], starts[b+1]) of X[:,f]
// ---------------------------------------------------------------------------
__global__ void pool_max_kernel(const float* __restrict__ X,
                                const int* __restrict__ starts, int F,
                                float* __restrict__ Y) {
    int b = blockIdx.x;
    int f = blockIdx.y * blockDim.x + threadIdx.x;
    if (f >= F) return;
    int s = starts[b], e = starts[b + 1];
    float m = -__builtin_inff();
    for (int r = s; r < e; ++r) m = fmaxf(m, X[(size_t)r * F + f]);
    Y[(size_t)b * F + f] = m;
}

// ---------------------------------------------------------------------------
// Final head: out[b] = dot(Xf[b,:256], out_w) + out_b
// ---------------------------------------------------------------------------
__global__ void out_kernel(const float* __restrict__ Xf, const float* __restrict__ out_w,
                           const float* __restrict__ out_b, float* __restrict__ out, int B) {
    int b = blockIdx.x * blockDim.x + threadIdx.x;
    if (b >= B) return;
    float acc = out_b[0];
#pragma unroll 8
    for (int k = 0; k < 256; ++k) acc += Xf[(size_t)b * 256 + k] * out_w[k];
    out[b] = acc;
}

// ---------------------------------------------------------------------------
extern "C" void kernel_launch(void* const* d_in, const int* in_sizes, int n_in,
                              void* d_out, int out_size, void* d_ws, size_t ws_size,
                              hipStream_t stream) {
    const float* seq_x      = (const float*)d_in[0];
    const float* compound_x = (const float*)d_in[1];
    const float* conv_w     = (const float*)d_in[2];
    const float* conv_b     = (const float*)d_in[3];
    const float* c1_wl = (const float*)d_in[4],  *c1_bl = (const float*)d_in[5],  *c1_wr = (const float*)d_in[6];
    const float* c2_wl = (const float*)d_in[7],  *c2_bl = (const float*)d_in[8],  *c2_wr = (const float*)d_in[9];
    const float* c3_wl = (const float*)d_in[10], *c3_bl = (const float*)d_in[11], *c3_wr = (const float*)d_in[12];
    const float* cfc_w = (const float*)d_in[13], *cfc_b = (const float*)d_in[14];
    const float* p1_wl = (const float*)d_in[15], *p1_bl = (const float*)d_in[16], *p1_wr = (const float*)d_in[17];
    const float* p2_wl = (const float*)d_in[18], *p2_bl = (const float*)d_in[19], *p2_wr = (const float*)d_in[20];
    const float* p3_wl = (const float*)d_in[21], *p3_bl = (const float*)d_in[22], *p3_wr = (const float*)d_in[23];
    const float* pfc_w = (const float*)d_in[24], *pfc_b = (const float*)d_in[25];
    const float* fc1_w = (const float*)d_in[26], *fc1_b = (const float*)d_in[27];
    const float* out_w = (const float*)d_in[28], *out_b = (const float*)d_in[29];
    const int* prot_gather    = (const int*)d_in[30];
    const int* prot_src       = (const int*)d_in[31];
    const int* prot_tgt       = (const int*)d_in[32];
    const int* comp_src       = (const int*)d_in[33];
    const int* comp_tgt       = (const int*)d_in[34];
    const int* protein_batch  = (const int*)d_in[35];
    const int* compound_batch = (const int*)d_in[36];

    const int NP = in_sizes[30];   // 49152
    const int Ep = in_sizes[31];   // 393216
    const int Ec = in_sizes[33];   // 10240
    const int NC = in_sizes[36];   // 2560
    const int B  = out_size;       // 64

    // ---- workspace carve-up ----
    float* ws = (float*)d_ws;
    size_t off = 0;
    float* prot_bufA = ws + off; off += (size_t)NP * 512;
    float* prot_bufB = ws + off; off += (size_t)NP * 512;
    float* prot_agg  = ws + off; off += (size_t)NP * 256;
    float* comp_xpad = ws + off; off += (size_t)NC * 96;
    float* comp_bufA = ws + off; off += (size_t)NC * 512;
    float* comp_bufB = ws + off; off += (size_t)NC * 512;
    float* comp_agg  = ws + off; off += (size_t)NC * 256;
    float* pool_c    = ws + off; off += (size_t)B * 512;
    float* pool_p    = ws + off; off += (size_t)B * 512;
    float* xcat      = ws + off; off += (size_t)B * 256;
    float* fc1o      = ws + off; off += (size_t)B * 256;
    int*   pstarts   = (int*)(ws + off); off += (size_t)(B + 2);
    int*   cstarts   = (int*)(ws + off); off += (size_t)(B + 2);
    off = (off + 7) & ~(size_t)7;                  // 32B align for v16h panels
    _Float16* hbase = (_Float16*)(ws + off);
    size_t hoff = 0;
    auto halloc = [&](size_t nh) { _Float16* p = hbase + hoff; hoff += (nh + 15) & ~(size_t)15; return p; };
    // weight panels
    _Float16* c1Pl = halloc((size_t)96 * 128),  *c1Pr = halloc((size_t)96 * 128);
    _Float16* c2Pl = halloc((size_t)128 * 256), *c2Pr = halloc((size_t)128 * 256);
    _Float16* c3Pl = halloc((size_t)256 * 512), *c3Pr = halloc((size_t)256 * 512);
    _Float16* p1Pl = halloc((size_t)128 * 128), *p1Pr = halloc((size_t)128 * 128);
    _Float16* p2Pl = halloc((size_t)128 * 256), *p2Pr = halloc((size_t)128 * 256);
    _Float16* p3Pl = halloc((size_t)256 * 512), *p3Pr = halloc((size_t)256 * 512);
    _Float16* cfcP = halloc((size_t)512 * 128);
    _Float16* pfcP = halloc((size_t)512 * 128);
    _Float16* fc1P = halloc((size_t)256 * 256);
    // activation panels (A-side), sized for the largest K used
    _Float16* prot_aggP = halloc((size_t)NP * 256);
    _Float16* prot_xP   = halloc((size_t)NP * 256);
    _Float16* comp_aggP = halloc((size_t)NC * 256);
    _Float16* comp_xP   = halloc((size_t)NC * 256);
    _Float16* headP     = halloc((size_t)B * 512);
    (void)ws_size; (void)n_in;

    auto packB = [&](const float* W, int Ksrc, int N, int Kpad, _Float16* P) {
        int total = Kpad * N;
        pack_b_kernel<<<(total + 255) / 256, 256, 0, stream>>>(W, Ksrc, N, Kpad, P);
    };
    auto packA = [&](const float* X, int M, int K, _Float16* P) {
        long long total = (long long)M * K;
        pack_a_kernel<<<(int)((total + 255) / 256), 256, 0, stream>>>(X, K, total, P);
    };
    auto scatter = [&](const float* X, const int* s, const int* t, int E, int F, int astride,
                       float* A) {
        long long total = (long long)E * F;
        scatter_add_kernel<<<(int)((total + 255) / 256), 256, 0, stream>>>(X, s, t, total, F,
                                                                           astride, A);
    };

    // ---- weight packing (every call; deterministic, ~1.8MB total) ----
    packB(c1_wl,  78, 128,  96, c1Pl);  packB(c1_wr,  78, 128,  96, c1Pr);
    packB(c2_wl, 128, 256, 128, c2Pl);  packB(c2_wr, 128, 256, 128, c2Pr);
    packB(c3_wl, 256, 512, 256, c3Pl);  packB(c3_wr, 256, 512, 256, c3Pr);
    packB(p1_wl, 128, 128, 128, p1Pl);  packB(p1_wr, 128, 128, 128, p1Pr);
    packB(p2_wl, 128, 256, 128, p2Pl);  packB(p2_wr, 128, 256, 128, p2Pr);
    packB(p3_wl, 256, 512, 256, p3Pl);  packB(p3_wr, 256, 512, 256, p3Pr);
    packB(cfc_w, 512, 128, 512, cfcP);
    packB(pfc_w, 512, 128, 512, pfcP);
    packB(fc1_w, 256, 256, 256, fc1P);

    // ---- segment starts + padded compound features ----
    seg_starts_kernel<<<1, 128, 0, stream>>>(protein_batch, NP, B, pstarts);
    seg_starts_kernel<<<1, 128, 0, stream>>>(compound_batch, NC, B, cstarts);
    {
        long long total = (long long)NC * 96;
        pad_copy_kernel<<<(int)((total + 255) / 256), 256, 0, stream>>>(compound_x, 78, 96,
                                                                        total, comp_xpad);
    }

    // conv1d + ragged gather -> p0 [NP,128] in prot_bufA
    conv_gather_kernel<<<NP, 128, 0, stream>>>(seq_x, conv_w, conv_b, prot_gather, prot_bufA);

    // ---- compound branch (K padded 78->96 in layer 1) ----
    hipMemsetAsync(comp_agg, 0, (size_t)NC * 96 * sizeof(float), stream);
    scatter(compound_x, comp_src, comp_tgt, Ec, 78, 96, comp_agg);
    packA(comp_agg, NC, 96, comp_aggP);
    packA(comp_xpad, NC, 96, comp_xP);
    wmma_sage_kernel<<<dim3(NC / 16, 128 / 64), 32, 0, stream>>>(
        comp_aggP, comp_xP, c1Pl, c1Pr, c1_bl, comp_bufA, NC, 96, 128);

    hipMemsetAsync(comp_agg, 0, (size_t)NC * 128 * sizeof(float), stream);
    scatter(comp_bufA, comp_src, comp_tgt, Ec, 128, 128, comp_agg);
    packA(comp_agg, NC, 128, comp_aggP);
    packA(comp_bufA, NC, 128, comp_xP);
    wmma_sage_kernel<<<dim3(NC / 16, 256 / 64), 32, 0, stream>>>(
        comp_aggP, comp_xP, c2Pl, c2Pr, c2_bl, comp_bufB, NC, 128, 256);

    hipMemsetAsync(comp_agg, 0, (size_t)NC * 256 * sizeof(float), stream);
    scatter(comp_bufB, comp_src, comp_tgt, Ec, 256, 256, comp_agg);
    packA(comp_agg, NC, 256, comp_aggP);
    packA(comp_bufB, NC, 256, comp_xP);
    wmma_sage_kernel<<<dim3(NC / 16, 512 / 64), 32, 0, stream>>>(
        comp_aggP, comp_xP, c3Pl, c3Pr, c3_bl, comp_bufA, NC, 256, 512);

    // ---- protein branch ----
    hipMemsetAsync(prot_agg, 0, (size_t)NP * 128 * sizeof(float), stream);
    scatter(prot_bufA, prot_src, prot_tgt, Ep, 128, 128, prot_agg);
    packA(prot_agg, NP, 128, prot_aggP);
    packA(prot_bufA, NP, 128, prot_xP);
    wmma_sage_kernel<<<dim3(NP / 16, 128 / 64), 32, 0, stream>>>(
        prot_aggP, prot_xP, p1Pl, p1Pr, p1_bl, prot_bufB, NP, 128, 128);

    hipMemsetAsync(prot_agg, 0, (size_t)NP * 128 * sizeof(float), stream);
    scatter(prot_bufB, prot_src, prot_tgt, Ep, 128, 128, prot_agg);
    packA(prot_agg, NP, 128, prot_aggP);
    packA(prot_bufB, NP, 128, prot_xP);
    wmma_sage_kernel<<<dim3(NP / 16, 256 / 64), 32, 0, stream>>>(
        prot_aggP, prot_xP, p2Pl, p2Pr, p2_bl, prot_bufA, NP, 128, 256);

    hipMemsetAsync(prot_agg, 0, (size_t)NP * 256 * sizeof(float), stream);
    scatter(prot_bufA, prot_src, prot_tgt, Ep, 256, 256, prot_agg);
    packA(prot_agg, NP, 256, prot_aggP);
    packA(prot_bufA, NP, 256, prot_xP);
    wmma_sage_kernel<<<dim3(NP / 16, 512 / 64), 32, 0, stream>>>(
        prot_aggP, prot_xP, p3Pl, p3Pr, p3_bl, prot_bufB, NP, 256, 512);

    // ---- pooling ----
    pool_max_kernel<<<dim3(B, 2), 256, 0, stream>>>(comp_bufA, cstarts, 512, pool_c);
    pool_max_kernel<<<dim3(B, 2), 256, 0, stream>>>(prot_bufB, pstarts, 512, pool_p);

    // ---- heads: cfc / pfc write the concat buffer directly ----
    packA(pool_c, B, 512, headP);
    wmma_linear_kernel<<<dim3(B / 16, 128 / 64), 32, 0, stream>>>(
        headP, cfcP, cfc_b, xcat, 256, 0,   B, 512, 128, 1);
    packA(pool_p, B, 512, headP);
    wmma_linear_kernel<<<dim3(B / 16, 128 / 64), 32, 0, stream>>>(
        headP, pfcP, pfc_b, xcat, 256, 128, B, 512, 128, 1);
    packA(xcat, B, 256, headP);
    wmma_linear_kernel<<<dim3(B / 16, 256 / 64), 32, 0, stream>>>(
        headP, fc1P, fc1_b, fc1o, 256, 0,   B, 256, 256, 1);

    out_kernel<<<1, 64, 0, stream>>>(fc1o, out_w, out_b, (float*)d_out, B);
}
